// MemristiveFullyConnected_49503793053702
// MI455X (gfx1250) — matrix operand order repaired and checked by reading, not compiled, not run on hardware
//
#include <hip/hip_runtime.h>

// ---------------------------------------------------------------------------
// y = x @ w + b   (the memristive conductance mapping cancels analytically)
//   x: (8192, 4096) f32   w: (4096, 4096) f32   b: (4096,) f32
// f16 hi/lo split-GEMM on v_wmma_f32_16x16x32_f16, fp32 accumulate.
// ---------------------------------------------------------------------------

typedef __attribute__((ext_vector_type(16))) _Float16 v16h;
typedef __attribute__((ext_vector_type(8)))  float    v8f;

#define BATCH 8192
#define NIN   4096
#define NOUT  4096
#define BM    128
#define BN    128
#define BK    32
#define LDSW  40   // f16 per LDS row: 32 data + 8 pad = 80B (16B aligned)

union FragU { uint4 q[2]; v16h v; };

// A fragment (16x32, M=lane&15): dwords {0..3}+4h and {8..11}+4h of the row
__device__ __forceinline__ v16h load_a_frag(const _Float16* rowBase, int h) {
  const uint4* q = (const uint4*)rowBase;
  FragU f; f.q[0] = q[h]; f.q[1] = q[2 + h];
  return f.v;
}
// B fragment (32x16, N=lane&15, row holds K-contiguous data): dwords {0..3}+4h and {8..11}+4h
__device__ __forceinline__ v16h load_b_frag(const _Float16* rowBase, int h) {
  const uint4* q = (const uint4*)rowBase;
  FragU f; f.q[0] = q[h]; f.q[1] = q[2 + h];
  return f.v;
}

__global__ __launch_bounds__(256) void memristive_fc_gemm(
    const float* __restrict__ X, const float* __restrict__ Wt,
    const float* __restrict__ Bias, float* __restrict__ Y) {
  __shared__ __align__(16) _Float16 aH[BM * LDSW];
  __shared__ __align__(16) _Float16 aL[BM * LDSW];
  __shared__ __align__(16) _Float16 bH[BN * LDSW];
  __shared__ __align__(16) _Float16 bL[BN * LDSW];

  const int tid  = threadIdx.x;
  const int lane = tid & 31;
  const int wave = tid >> 5;
  const int wm   = wave >> 1;   // 0..3 : 32-row slice
  const int wn   = wave & 1;    // 0..1 : 64-col slice
  const int l16  = lane & 15;
  const int hh   = lane >> 4;

  const int n0 = blockIdx.x * BN;
  const int m0 = blockIdx.y * BM;

  v8f acc[2][4];
  const v8f vzero = {0.f, 0.f, 0.f, 0.f, 0.f, 0.f, 0.f, 0.f};
#pragma unroll
  for (int mi = 0; mi < 2; ++mi)
#pragma unroll
    for (int ni = 0; ni < 4; ++ni) acc[mi][ni] = vzero;

  for (int k0 = 0; k0 < NIN; k0 += BK) {
    __syncthreads();

    // ---- stage A tile (BM x BK), fp32 -> f16 hi/lo, row-major along K ----
#pragma unroll
    for (int j = 0; j < 8; ++j) {
      int p   = tid + 256 * j;          // 0..2047 float2 pairs
      int row = p >> 4;
      int kp  = p & 15;
      const float2 xv =
          *(const float2*)(X + (size_t)(m0 + row) * NIN + (k0 + 2 * kp));
      _Float16 h0 = (_Float16)xv.x;
      _Float16 h1 = (_Float16)xv.y;
      _Float16 l0 = (_Float16)(xv.x - (float)h0);
      _Float16 l1 = (_Float16)(xv.y - (float)h1);
      int idx = row * LDSW + 2 * kp;
      aH[idx] = h0; aH[idx + 1] = h1;
      aL[idx] = l0; aL[idx + 1] = l1;
    }

    // ---- stage B tile (BK x BN) transposed -> LDS[n][k], f16 hi/lo ----
#pragma unroll
    for (int j = 0; j < 16; ++j) {
      int p  = tid + 256 * j;           // 0..4095 elements
      int nn = p & (BN - 1);
      int kk = p >> 7;
      float wv = Wt[(size_t)(k0 + kk) * NOUT + (n0 + nn)];
      _Float16 h0 = (_Float16)wv;
      _Float16 l0 = (_Float16)(wv - (float)h0);
      bH[nn * LDSW + kk] = h0;
      bL[nn * LDSW + kk] = l0;
    }

    // warm L2 for the next K tile (global_prefetch_b8)
    if (k0 + BK < NIN) {
      __builtin_prefetch(X + (size_t)(m0 + (tid >> 1)) * NIN + (k0 + BK), 0, 1);
      __builtin_prefetch(Wt + (size_t)(k0 + BK + (tid >> 4)) * NOUT + n0, 0, 1);
    }

    __syncthreads();

    // ---- load fragments from LDS ----
    v16h afH[2], afL[2], bfH[4], bfL[4];
#pragma unroll
    for (int mi = 0; mi < 2; ++mi) {
      int arow = (wm * 32 + mi * 16 + l16) * LDSW;
      afH[mi] = load_a_frag(&aH[arow], hh);
      afL[mi] = load_a_frag(&aL[arow], hh);
    }
#pragma unroll
    for (int ni = 0; ni < 4; ++ni) {
      int brow = (wn * 64 + ni * 16 + l16) * LDSW;
      bfH[ni] = load_b_frag(&bH[brow], hh);
      bfL[ni] = load_b_frag(&bL[brow], hh);
    }

    // ---- 3-pass compensated f16 WMMA: hi*hi + hi*lo + lo*hi ----
#pragma unroll
    for (int mi = 0; mi < 2; ++mi)
#pragma unroll
      for (int ni = 0; ni < 4; ++ni) {
        acc[mi][ni] = __builtin_amdgcn_wmma_f32_16x16x32_f16(
            false, afH[mi], false, bfH[ni], (short)0, acc[mi][ni], false, false);
        acc[mi][ni] = __builtin_amdgcn_wmma_f32_16x16x32_f16(
            false, afH[mi], false, bfL[ni], (short)0, acc[mi][ni], false, false);
        acc[mi][ni] = __builtin_amdgcn_wmma_f32_16x16x32_f16(
            false, afL[mi], false, bfH[ni], (short)0, acc[mi][ni], false, false);
      }
  }

  // ---- epilogue: add bias, store fp32 ----
#pragma unroll
  for (int mi = 0; mi < 2; ++mi)
#pragma unroll
    for (int ni = 0; ni < 4; ++ni) {
      int col   = n0 + wn * 64 + ni * 16 + l16;
      float bv  = Bias[col];
      int rbase = m0 + wm * 32 + mi * 16 + hh * 8;
#pragma unroll
      for (int i = 0; i < 8; ++i) {
        Y[(size_t)(rbase + i) * NOUT + col] = acc[mi][ni][i] + bv;
      }
    }
}

extern "C" void kernel_launch(void* const* d_in, const int* in_sizes, int n_in,
                              void* d_out, int out_size, void* d_ws,
                              size_t ws_size, hipStream_t stream) {
  (void)in_sizes; (void)n_in; (void)out_size; (void)d_ws; (void)ws_size;
  const float* x = (const float*)d_in[0];
  const float* w = (const float*)d_in[1];
  const float* b = (const float*)d_in[2];
  float* y = (float*)d_out;
  dim3 grid(NOUT / BN, BATCH / BM);  // (32, 64), exact cover -> EXEC all-ones
  memristive_fc_gemm<<<grid, dim3(256), 0, stream>>>(x, w, b, y);
}
